// DirectedClassifier_71133248356971
// MI455X (gfx1250) — matrix operand order repaired and checked by reading, not compiled, not run on hardware
//
#include <hip/hip_runtime.h>
#include <hip/hip_bf16.h>

// ---------------- problem constants ----------------
#define B_   16
#define L_   1024
#define DIN_ 512
#define H_   8
#define DK_  64
#define NROW (B_*L_)                 // 16384
#define PLANE ((size_t)NROW*DIN_)    // 8,388,608 elems

// ---------------- WMMA vector types ----------------
typedef __bf16 bf16x16 __attribute__((ext_vector_type(16)));
typedef __bf16 bf16x8  __attribute__((ext_vector_type(8)));
typedef float  v8f     __attribute__((ext_vector_type(8)));
typedef __bf16 bf16x2  __attribute__((ext_vector_type(2)));

// ---------------- TDM descriptor types ----------------
typedef unsigned int uint32x4 __attribute__((ext_vector_type(4)));
typedef int          int32x8  __attribute__((ext_vector_type(8)));
typedef int          int32x4  __attribute__((ext_vector_type(4)));

#if defined(__has_builtin)
#  if __has_builtin(__builtin_amdgcn_tensor_load_to_lds) && \
      __has_builtin(__builtin_amdgcn_s_wait_tensorcnt)
#    define USE_TDM 1
#  endif
#endif
#ifndef USE_TDM
#  define USE_TDM 0
#endif

union AFrag { bf16x16 v; bf16x8 h[2]; };

// =====================================================================
// LayerNorm: one block per row, grid.y selects q/k/v plane. bf16 output.
// =====================================================================
__global__ __launch_bounds__(256) void ln_kernel(
    const float* __restrict__ q, const float* __restrict__ k, const float* __restrict__ v,
    const float* __restrict__ g1, const float* __restrict__ b1,
    const float* __restrict__ g2, const float* __restrict__ b2,
    const float* __restrict__ g3, const float* __restrict__ b3,
    __bf16* __restrict__ xn)
{
    __shared__ float red[16];
    const int row = blockIdx.x, plane = blockIdx.y, t = threadIdx.x;
    const float* src = plane == 0 ? q : (plane == 1 ? k : v);
    const float* gg  = plane == 0 ? g1 : (plane == 1 ? g2 : g3);
    const float* bb  = plane == 0 ? b1 : (plane == 1 ? b2 : b3);

    float2 xv = ((const float2*)(src + (size_t)row * DIN_))[t];
    float s  = xv.x + xv.y;
    float s2 = xv.x * xv.x + xv.y * xv.y;
    for (int off = 16; off; off >>= 1) {
        s  += __shfl_xor(s,  off, 32);
        s2 += __shfl_xor(s2, off, 32);
    }
    const int w = t >> 5, lane = t & 31;
    if (lane == 0) { red[w] = s; red[8 + w] = s2; }
    __syncthreads();
    if (w == 0) {
        float a  = (lane < 8) ? red[lane]     : 0.f;
        float bx = (lane < 8) ? red[8 + lane] : 0.f;
        for (int off = 4; off; off >>= 1) {
            a  += __shfl_xor(a,  off, 32);
            bx += __shfl_xor(bx, off, 32);
        }
        if (lane == 0) { red[0] = a; red[1] = bx; }
    }
    __syncthreads();
    const float mu  = red[0] * (1.f / DIN_);
    const float var = red[1] * (1.f / DIN_) - mu * mu;
    const float rs  = rsqrtf(var + 1e-5f);

    float y0 = (xv.x - mu) * rs * gg[2 * t]     + bb[2 * t];
    float y1 = (xv.y - mu) * rs * gg[2 * t + 1] + bb[2 * t + 1];
    bf16x2 out; out[0] = (__bf16)y0; out[1] = (__bf16)y1;
    *(bf16x2*)(xn + (size_t)plane * PLANE + (size_t)row * DIN_ + 2 * t) = out;
}

// =====================================================================
// f32 -> bf16 weight conversion
// =====================================================================
__global__ __launch_bounds__(256) void cvt_kernel(const float* __restrict__ src,
                                                  __bf16* __restrict__ dst, int n)
{
    int i = blockIdx.x * 256 + threadIdx.x;
    if (i < n) dst[i] = (__bf16)src[i];
}

// =====================================================================
// NT GEMM: C[M,512] = A[M,512] * W[512,512]^T   (K = N = 512 fixed)
// block tile 64(M) x 128(N); 8 waves, each 16(M) x 64(N) via 4 WMMA accs.
// =====================================================================
template <bool OUT_BF16>
__global__ __launch_bounds__(256) void gemm512_nt(const __bf16* __restrict__ A,
                                                  const __bf16* __restrict__ W,
                                                  void* __restrict__ Cout)
{
    __shared__ __bf16 As[64 * 32];
    __shared__ __bf16 Bs[128 * 32];
    const int t = threadIdx.x, lane = t & 31, w = t >> 5, ln = lane & 15;
    const int m0 = blockIdx.x * 64, n0 = blockIdx.y * 128;
    const int mw = w & 3, nw = w >> 2;

    v8f acc[4] = {};
    for (int k0 = 0; k0 < 512; k0 += 32) {
        __syncthreads();
        {   // A tile: 2048 bf16, 8 per thread
            int r = t >> 2, cg = t & 3;
            *(bf16x8*)&As[r * 32 + cg * 8] =
                *(const bf16x8*)&A[((size_t)(m0 + r)) * 512 + k0 + cg * 8];
        }
        for (int i = 0; i < 2; ++i) {   // B tile: 4096 bf16, 16 per thread
            int idx = t + i * 256, r = idx >> 2, cg = idx & 3;
            *(bf16x8*)&Bs[r * 32 + cg * 8] =
                *(const bf16x8*)&W[((size_t)(n0 + r)) * 512 + k0 + cg * 8];
        }
        __syncthreads();

        AFrag a;    // A layout: lanes 0-15 -> k 0..7 / 16..23, lanes 16-31 -> 8..15 / 24..31
        const int ab = (mw * 16 + ln) * 32 + ((lane < 16) ? 0 : 8);
        a.h[0] = *(bf16x8*)&As[ab];
        a.h[1] = *(bf16x8*)&As[ab + 16];
        #pragma unroll
        for (int j = 0; j < 4; ++j) {
            // B layout: lane column = ln, lanes 0-15 -> k 0..7 / 16..23, lanes 16-31 -> 8..15 / 24..31
            AFrag bfr;
            const int bb = (nw * 64 + j * 16 + ln) * 32 + ((lane < 16) ? 0 : 8);
            bfr.h[0] = *(bf16x8*)&Bs[bb];
            bfr.h[1] = *(bf16x8*)&Bs[bb + 16];
            bf16x16 bv = bfr.v;
            acc[j] = __builtin_amdgcn_wmma_f32_16x16x32_bf16(
                false, a.v, false, bv, (short)0, acc[j], false, false);
        }
    }
    const int mhi = (lane >= 16) ? 8 : 0;
    #pragma unroll
    for (int j = 0; j < 4; ++j)
        #pragma unroll
        for (int i = 0; i < 8; ++i) {
            size_t row = (size_t)(m0 + mw * 16 + i + mhi);
            size_t col = (size_t)(n0 + nw * 64 + j * 16 + ln);
            if (OUT_BF16) ((__bf16*)Cout)[row * 512 + col] = (__bf16)acc[j][i];
            else          ((float*)Cout)[row * 512 + col]  = acc[j][i];
        }
}

// =====================================================================
// Fused attention: per (q-tile 32, head, batch).
//   Phase 1: S = mask(Q K^T / 8)   (scores strip in dynamic LDS)
//            K chunks staged via TDM (tensor_load_to_lds) when available
//   softmax (one wave per row), write attn (f32) once to HBM
//   Phase 2: ctx = P V  (V staged transposed in LDS, manual: needs transpose)
// =====================================================================
#define SST 1040   // f32 stride of S rows (padded)
#define VST 136    // bf16 stride of transposed-V rows (272B, 16B aligned)
#define KV_LDS_OFF (32u * SST * 4u + 32u * 64u * 2u)   // byte offset of KV buffer

__global__ __launch_bounds__(256) void attn_kernel(
    const __bf16* __restrict__ Qp, const __bf16* __restrict__ Kp,
    const __bf16* __restrict__ Vp, const int* __restrict__ mask,
    float* __restrict__ attn_out, __bf16* __restrict__ ctx)
{
    extern __shared__ char smem[];
    float*  S  = (float*)smem;                               // [32][SST]
    __bf16* Qt = (__bf16*)(smem + 32 * SST * 4);             // [32][64]
    __bf16* KV = Qt + 32 * 64;                               // Kc[128][64] / Vt[64][VST]

    const int t = threadIdx.x, lane = t & 31, w = t >> 5, ln = lane & 15;
    const int h = blockIdx.y, b = blockIdx.z;
    const int q0 = blockIdx.x * 32;
    const size_t rowQ = (size_t)b * L_ + q0;

    {   // Q tile: 32x64 bf16
        int r = t >> 3, cg = t & 7;
        *(bf16x8*)&Qt[r * 64 + cg * 8] =
            *(const bf16x8*)&Qp[(rowQ + r) * 512 + h * 64 + cg * 8];
    }

    // ---------------- Phase 1: scores ----------------
    for (int kc = 0; kc < 8; ++kc) {
        __syncthreads();   // prior readers of KV are done
#if USE_TDM
        if (w == 0) {
            // D# for a 128x64 2-byte tile: row stride 512 elems, dest = KV (contiguous)
            const unsigned long long ga =
                (unsigned long long)(const void*)&Kp[((size_t)b * L_ + kc * 128) * 512 + h * 64];
            uint32x4 g0;
            g0[0] = 1u;                                   // count=1, user mode
            g0[1] = KV_LDS_OFF;                           // lds_addr (bytes)
            g0[2] = (unsigned int)(ga & 0xffffffffu);     // global_addr[31:0]
            g0[3] = (unsigned int)(ga >> 32) | (2u << 30);// global_addr[56:32] | type=2
            int32x8 g1 = {};
            g1[0] = (int)(1u << 16);                      // data_size=1 (2 bytes), wg_mask=0
            g1[1] = (int)(512u << 16);                    // tensor_dim0 = 512 (low 16)
            g1[2] = 0;                                    // tensor_dim0 hi=0, tensor_dim1 lo=0
            g1[3] = (int)(16u | (64u << 16));             // tensor_dim1 hi (1M rows), tile_dim0=64
            g1[4] = 128;                                  // tile_dim1=128, tile_dim2=0
            g1[5] = 512;                                  // tensor_dim0_stride = 512
            g1[6] = (int)(512u << 16);                    // stride0 hi=0 | tensor_dim1_stride lo
            g1[7] = 0;
            int32x4 z4 = {};
            int32x8 z8 = {};
            __builtin_amdgcn_tensor_load_to_lds(g0, g1, z4, z4, z8, 0);
            __builtin_amdgcn_s_wait_tensorcnt(0);
        }
#else
        #pragma unroll
        for (int i = 0; i < 4; ++i) {       // K chunk 128x64 bf16 (manual fallback)
            int idx = t + i * 256, r = idx >> 3, cg = idx & 7;
            *(bf16x8*)&KV[r * 64 + cg * 8] =
                *(const bf16x8*)&Kp[((size_t)b * L_ + kc * 128 + r) * 512 + h * 64 + cg * 8];
        }
#endif
        if (kc < 7)   // pull next chunk toward L2 (global_prefetch_b8)
            __builtin_prefetch(&Kp[((size_t)b * L_ + (kc + 1) * 128 + (t >> 1)) * 512 + h * 64], 0, 1);
        __syncthreads();

        #pragma unroll
        for (int qs = 0; qs < 2; ++qs) {
            v8f acc = {};
            #pragma unroll
            for (int ks = 0; ks < 2; ++ks) {
                AFrag a;
                int ab = (qs * 16 + ln) * 64 + ks * 32 + ((lane < 16) ? 0 : 8);
                a.h[0] = *(bf16x8*)&Qt[ab];
                a.h[1] = *(bf16x8*)&Qt[ab + 16];
                AFrag bfr;
                int bb = (w * 16 + ln) * 64 + ks * 32 + ((lane < 16) ? 0 : 8);
                bfr.h[0] = *(bf16x8*)&KV[bb];
                bfr.h[1] = *(bf16x8*)&KV[bb + 16];
                bf16x16 bv = bfr.v;
                acc = __builtin_amdgcn_wmma_f32_16x16x32_bf16(
                    false, a.v, false, bv, (short)0, acc, false, false);
            }
            const int kg  = kc * 128 + w * 16 + ln;      // key column (C-layout: N = lane%16)
            const int mhi = (lane >= 16) ? 8 : 0;
            #pragma unroll
            for (int i = 0; i < 8; ++i) {
                int ql = qs * 16 + i + mhi;
                int qg = q0 + ql;
                int mk = mask[((size_t)b * L_ + qg) * L_ + kg];
                bool ok = (qg != kg) && (mk == 0);       // diag & pad mask
                S[ql * SST + kg] = ok ? acc[i] * 0.125f : -1.0e30f;
            }
        }
    }
    __syncthreads();

    // ---------------- softmax (wave per row) ----------------
    for (int tr = 0; tr < 4; ++tr) {
        const int r = w + tr * 8, qg = q0 + r;
        float m = -3.0e38f;
        for (int c = lane; c < L_; c += 32) m = fmaxf(m, S[r * SST + c]);
        for (int off = 16; off; off >>= 1) m = fmaxf(m, __shfl_xor(m, off, 32));
        float sum = 0.f;
        for (int c = lane; c < L_; c += 32) {
            float v = S[r * SST + c];
            float p = (v <= -1.0e29f) ? 0.f : __expf(v - m);   // masked -> exactly 0
            S[r * SST + c] = p;
            sum += p;
        }
        for (int off = 16; off; off >>= 1) sum += __shfl_xor(sum, off, 32);
        const float inv = 1.0f / (sum + 1e-13f);
        const size_t arow = (((size_t)(h * B_ + b)) * L_ + qg) * L_;   // head-major out
        for (int c = lane; c < L_; c += 32) {
            float p = S[r * SST + c] * inv;
            S[r * SST + c] = p;
            attn_out[arow + c] = p;
        }
    }

    // ---------------- Phase 2: ctx = P @ V ----------------
    const int qs = w >> 2, ds = w & 3;
    v8f o = {};
    for (int kc = 0; kc < 8; ++kc) {
        __syncthreads();
        {   // stage V chunk transposed: Vt[d][kk]
            int r = t >> 1, half = t & 1;
            const __bf16* vsrc = &Vp[((size_t)b * L_ + kc * 128 + r) * 512 + h * 64 + half * 32];
            bf16x8 vv[4];
            #pragma unroll
            for (int p = 0; p < 4; ++p) vv[p] = *(const bf16x8*)&vsrc[p * 8];
            #pragma unroll
            for (int p = 0; p < 4; ++p)
                #pragma unroll
                for (int j = 0; j < 8; ++j)
                    KV[(half * 32 + p * 8 + j) * VST + r] = vv[p][j];
        }
        __syncthreads();
        #pragma unroll
        for (int ks = 0; ks < 4; ++ks) {
            AFrag a;
            const int ar = (qs * 16 + ln) * SST + kc * 128 + ks * 32;
            const int kb = (lane < 16) ? 0 : 8;
            #pragma unroll
            for (int j = 0; j < 8; ++j) a.v[j]     = (__bf16)S[ar + kb + j];
            #pragma unroll
            for (int j = 0; j < 8; ++j) a.v[8 + j] = (__bf16)S[ar + 16 + kb + j];
            AFrag bfr;
            const int bb = (ds * 16 + ln) * VST + ks * 32 + kb;
            bfr.h[0] = *(bf16x8*)&KV[bb];
            bfr.h[1] = *(bf16x8*)&KV[bb + 16];
            bf16x16 bv = bfr.v;
            o = __builtin_amdgcn_wmma_f32_16x16x32_bf16(
                false, a.v, false, bv, (short)0, o, false, false);
        }
    }
    const int mhi = (lane >= 16) ? 8 : 0;
    #pragma unroll
    for (int i = 0; i < 8; ++i) {
        size_t row = rowQ + qs * 16 + i + mhi;
        ctx[row * 512 + h * 64 + ds * 16 + ln] = (__bf16)o[i];
    }
}

// =====================================================================
// launch
// =====================================================================
extern "C" void kernel_launch(void* const* d_in, const int* in_sizes, int n_in,
                              void* d_out, int out_size, void* d_ws, size_t ws_size,
                              hipStream_t stream)
{
    const float* q    = (const float*)d_in[0];
    const float* k    = (const float*)d_in[1];
    const float* v    = (const float*)d_in[2];
    const int*   mask = (const int*)  d_in[3];
    const float* Wq   = (const float*)d_in[4];
    const float* Wk   = (const float*)d_in[5];
    const float* Wv   = (const float*)d_in[6];
    const float* Wfc  = (const float*)d_in[7];
    const float* g1 = (const float*)d_in[8],  *b1 = (const float*)d_in[9];
    const float* g2 = (const float*)d_in[10], *b2 = (const float*)d_in[11];
    const float* g3 = (const float*)d_in[12], *b3 = (const float*)d_in[13];

    // workspace layout (bf16), ~114 MiB total
    __bf16* xn   = (__bf16*)d_ws;                 // 3 normalized planes
    __bf16* Wqb  = xn  + 3 * PLANE;
    __bf16* Wkb  = Wqb + 512 * 512;
    __bf16* Wvb  = Wkb + 512 * 512;
    __bf16* Wfcb = Wvb + 512 * 512;
    __bf16* Qp   = Wfcb + 512 * 512;              // [16384,512] col = h*64+d
    __bf16* Kp   = Qp + PLANE;
    __bf16* Vp   = Kp + PLANE;
    __bf16* ctx  = Vp + PLANE;

    float* out_dyn  = (float*)d_out;              // [16,1024,512]
    float* out_attn = out_dyn + PLANE;            // [128,1024,1024]

    // 1) LayerNorms (q,k,v)
    ln_kernel<<<dim3(NROW, 3), 256, 0, stream>>>(q, k, v, g1, b1, g2, b2, g3, b3, xn);

    // 2) weight conversion
    cvt_kernel<<<1024, 256, 0, stream>>>(Wq,  Wqb,  512 * 512);
    cvt_kernel<<<1024, 256, 0, stream>>>(Wk,  Wkb,  512 * 512);
    cvt_kernel<<<1024, 256, 0, stream>>>(Wv,  Wvb,  512 * 512);
    cvt_kernel<<<1024, 256, 0, stream>>>(Wfc, Wfcb, 512 * 512);

    // 3) projections (WMMA, bf16 out)
    gemm512_nt<true><<<dim3(NROW / 64, 4), 256, 0, stream>>>(xn,             Wqb, Qp);
    gemm512_nt<true><<<dim3(NROW / 64, 4), 256, 0, stream>>>(xn + PLANE,     Wkb, Kp);
    gemm512_nt<true><<<dim3(NROW / 64, 4), 256, 0, stream>>>(xn + 2 * PLANE, Wvb, Vp);

    // 4) fused attention (dynamic LDS ~151 KB, within 320 KB/WGP)
    size_t smem = (size_t)32 * SST * 4 + 32 * 64 * 2 + 64 * VST * 2;
    (void)hipFuncSetAttribute((const void*)attn_kernel,
                              hipFuncAttributeMaxDynamicSharedMemorySize, (int)smem);
    attn_kernel<<<dim3(L_ / 32, H_, B_), 256, smem, stream>>>(Qp, Kp, Vp, mask, out_attn, ctx);

    // 5) final FC (WMMA, f32 out)
    gemm512_nt<false><<<dim3(NROW / 64, 4), 256, 0, stream>>>(ctx, Wfcb, (void*)out_dyn);
}
